// Predictor_13099650252895
// MI455X (gfx1250) — matrix operand order repaired and checked by reading, not compiled, not run on hardware
//
#include <hip/hip_runtime.h>
#include <hip/hip_bf16.h>
#include <stdint.h>

// Sizes from the reference
#define B_     256
#define SEC_IN 800
#define NI     1000
#define NIP    1024   // NI padded to multiple of 32 (zero-filled)
#define H1N    256
#define H2N    16
#define PHN    256

#define IT 64   // interval tile per workgroup
#define KT 32   // K step (bf16 WMMA K)

typedef __attribute__((ext_vector_type(16))) __bf16 v16bf;
typedef __attribute__((ext_vector_type(2)))  __bf16 bf16x2;
typedef __attribute__((ext_vector_type(8)))  float  v8f;

union FragU {
    uint4 q[2];
    v16bf v;
};

__device__ __forceinline__ float relu(float x) { return x > 0.0f ? x : 0.0f; }

// -----------------------------------------------------------------------------
// Kernel 0: f32 -> bf16 weight conversion with optional zero-padded row stride.
// -----------------------------------------------------------------------------
__global__ __launch_bounds__(256) void cvt_pad_kernel(
    const float* __restrict__ src, __bf16* __restrict__ dst,
    int cols, int dstride, int total)
{
    int idx = blockIdx.x * 256 + threadIdx.x;
    if (idx >= total) return;
    int r = idx / dstride;
    int c = idx - r * dstride;
    dst[idx] = (c < cols) ? (__bf16)src[(size_t)r * cols + c] : (__bf16)0.0f;
}

// -----------------------------------------------------------------------------
// Kernel 1: fused section MLP.
// grid = (256 batches, 16 interval-tiles), block = 256 threads (8 waves).
// -----------------------------------------------------------------------------
__global__ __launch_bounds__(256) void sec_mlp_kernel(
    const float*  __restrict__ state,
    const __bf16* __restrict__ w1b,   // [H1N][SEC_IN] bf16
    const float*  __restrict__ b1,
    const __bf16* __restrict__ w2b,   // [H2N][H1N] bf16
    const float*  __restrict__ b2,
    const float*  __restrict__ W3, const float* __restrict__ b3,
    __bf16* __restrict__ fcb)         // [B_][NIP] bf16, zero-padded cols
{
    __shared__ float  stage[2][KT][IT];   // async f32 staging (16 KB)
    __shared__ __bf16 bt[IT][40];         // bf16 B-tile, transposed [i][k] (+pad)
    __shared__ __bf16 h1t[IT][264];       // bf16 H1, transposed [i][h] (+pad)

    const int b    = blockIdx.x;
    const int i0   = blockIdx.y * IT;
    const int tid  = threadIdx.x;
    const int lane = tid & 31;
    const int wave = tid >> 5;
    const int g    = lane >> 4;   // lane half (0/1)
    const int ln   = lane & 15;

    const long sbase  = (long)b * (SEC_IN * (long)NI);
    const long MAXOFF = (long)B_ * SEC_IN * NI - 4;

    // Issue async global->LDS copy of a 32x64 f32 tile (double-buffered).
    auto issue_tile = [&](int buf, int k0) {
        #pragma unroll
        for (int r = 0; r < 2; ++r) {
            int  f   = tid + r * 256;        // 512 float4s total
            int  row = f >> 4;               // 0..31 (k)
            int  c4  = f & 15;               // 0..15 (i/4)
            long gidx = sbase + (long)(k0 + row) * NI + (i0 + c4 * 4);
            if (gidx > MAXOFF) gidx = MAXOFF;           // clamp OOB tail
            unsigned goff  = (unsigned)(gidx * 4);
            unsigned laddr = (unsigned)(uintptr_t)&stage[buf][row][c4 * 4];
            asm volatile("global_load_async_to_lds_b128 %0, %1, %2"
                         :: "v"(laddr), "v"(goff), "s"(state) : "memory");
        }
    };

    // Per-thread unit for the transpose/convert pass: one k-pair x 4 intervals.
    const int k2  = tid >> 4;    // 0..15  (k-pair index)
    const int ic4 = tid & 15;    // 0..15  (interval group of 4)

    v8f acc[2][4] = {};   // wave owns 32 (M=h1) x 64 (N=i) of H1

    issue_tile(0, 0);
    const int NKSTEP = SEC_IN / KT;   // 25
    for (int kk = 0; kk < NKSTEP; ++kk) {
        const int buf = kk & 1;
        asm volatile("s_wait_asynccnt 0x0" ::: "memory");
        __syncthreads();                                    // tile kk visible
        if (kk + 1 < NKSTEP) issue_tile(buf ^ 1, (kk + 1) * KT);

        // convert f32 stage -> bf16 transposed bt[i][k]:
        // two b128 row reads (immediate-offset pair), 4 packed converts,
        // 4 b32 stores at constant immediate offsets.
        {
            const float4 r0 = *(const float4*)&stage[buf][2 * k2 + 0][4 * ic4];
            const float4 r1 = *(const float4*)&stage[buf][2 * k2 + 1][4 * ic4];
            __bf16* bp0 = &bt[4 * ic4][2 * k2];
            bf16x2 p0; p0[0] = (__bf16)r0.x; p0[1] = (__bf16)r1.x;
            bf16x2 p1; p1[0] = (__bf16)r0.y; p1[1] = (__bf16)r1.y;
            bf16x2 p2; p2[0] = (__bf16)r0.z; p2[1] = (__bf16)r1.z;
            bf16x2 p3; p3[0] = (__bf16)r0.w; p3[1] = (__bf16)r1.w;
            *(bf16x2*)(bp0)       = p0;
            *(bf16x2*)(bp0 + 40)  = p1;   // +1 interval row (80 B)
            *(bf16x2*)(bp0 + 80)  = p2;
            *(bf16x2*)(bp0 + 120) = p3;
        }
        __syncthreads();

        // B fragments: per-lane 16 contiguous K from LDS
        FragU bfrag[4];
        #pragma unroll
        for (int nf = 0; nf < 4; ++nf) {
            const uint4* p = (const uint4*)&bt[nf * 16 + ln][g * 16];
            bfrag[nf].q[0] = p[0];
            bfrag[nf].q[1] = p[1];
        }

        // A fragments: direct bf16 loads from pre-converted W1 (L2-resident)
        const int k0 = kk * KT;
        #pragma unroll
        for (int mf = 0; mf < 2; ++mf) {
            int m = wave * 32 + mf * 16 + ln;
            const __bf16* wrow = w1b + (size_t)m * SEC_IN + k0;
            FragU af;
            af.q[0] = *(const uint4*)(wrow + g * 16);       // K = k0+g*16+{0..7}
            af.q[1] = *(const uint4*)(wrow + g * 16 + 8);   // K = k0+g*16+8+{0..7}
            #pragma unroll
            for (int nf = 0; nf < 4; ++nf)
                acc[mf][nf] = __builtin_amdgcn_wmma_f32_16x16x32_bf16(
                    false, af.v, false, bfrag[nf].v, (short)0, acc[mf][nf], false, false);
        }
        // next iteration's wait+barrier covers WAR on bt/stage
    }

    // H1 epilogue: bias + relu, store bf16 transposed h1t[i][h]
    #pragma unroll
    for (int mf = 0; mf < 2; ++mf) {
        #pragma unroll
        for (int v = 0; v < 8; ++v) {
            int   m    = wave * 32 + mf * 16 + v + 8 * g;
            float bias = b1[m];
            #pragma unroll
            for (int nf = 0; nf < 4; ++nf)
                h1t[nf * 16 + ln][m] = (__bf16)relu(acc[mf][nf][v] + bias);
        }
    }
    __syncthreads();

    // Stage 2: H2 = relu(W2 x H1 + b2); sec = W3 . H2 + b3.  Waves 0..3.
    if (wave < 4) {
        const int nf = wave;
        v8f a2 = {};
        #pragma unroll
        for (int kk = 0; kk < H1N / KT; ++kk) {
            const int k0 = kk * KT;
            const __bf16* wrow = w2b + (size_t)ln * H1N + k0;
            FragU af;
            af.q[0] = *(const uint4*)(wrow + g * 16);
            af.q[1] = *(const uint4*)(wrow + g * 16 + 8);
            FragU bf;
            const uint4* p = (const uint4*)&h1t[nf * 16 + ln][k0 + g * 16];
            bf.q[0] = p[0];
            bf.q[1] = p[1];
            a2 = __builtin_amdgcn_wmma_f32_16x16x32_bf16(
                false, af.v, false, bf.v, (short)0, a2, false, false);
        }
        float partial = 0.0f;
        #pragma unroll
        for (int v = 0; v < 8; ++v) {
            int   m  = v + 8 * g;
            float hv = relu(a2[v] + b2[m]);
            partial += hv * W3[m];
        }
        float total = partial + __shfl_xor(partial, 16, 32) + b3[0];
        if (g == 0) {
            int i = i0 + nf * 16 + ln;                  // covers 0..1023 over blocks
            fcb[(size_t)b * NIP + i] = (__bf16)(i < NI ? total : 0.0f);
        }
    }
}

// -----------------------------------------------------------------------------
// Kernel 2: y = relu(fc @ Wp^T + bp).  One wave per 16x16 tile; both operands
// pre-converted bf16 with zero-padded K=1024 -> guard-free b128 loads.
// -----------------------------------------------------------------------------
__global__ __launch_bounds__(32) void pred_y_kernel(
    const __bf16* __restrict__ fcb,   // [B_][NIP]
    const __bf16* __restrict__ wpb,   // [PHN][NIP]
    const float* __restrict__ bp, float* __restrict__ y)
{
    const int lane = threadIdx.x & 31;
    const int g    = lane >> 4;
    const int ln   = lane & 15;
    const int m0   = blockIdx.x * 16;   // batch rows
    const int n0   = blockIdx.y * 16;   // PRED_H cols

    const __bf16* arow = fcb + (size_t)(m0 + ln) * NIP;
    const __bf16* brow = wpb + (size_t)(n0 + ln) * NIP;

    v8f acc = {};
    for (int kk = 0; kk < NIP / KT; ++kk) {
        const int k0 = kk * KT;
        FragU af, bf;
        af.q[0] = *(const uint4*)(arow + k0 + g * 16);
        af.q[1] = *(const uint4*)(arow + k0 + g * 16 + 8);
        const uint4* p = (const uint4*)(brow + k0 + g * 16);
        bf.q[0] = p[0];
        bf.q[1] = p[1];
        acc = __builtin_amdgcn_wmma_f32_16x16x32_bf16(
            false, af.v, false, bf.v, (short)0, acc, false, false);
    }
    #pragma unroll
    for (int v = 0; v < 8; ++v) {
        int m = m0 + v + 8 * g;
        int n = n0 + ln;
        y[(size_t)m * PHN + n] = relu(acc[v] + bp[n]);
    }
}

// -----------------------------------------------------------------------------
// Kernel 3: val[b] = y[b,:] . Wv + bv
// -----------------------------------------------------------------------------
__global__ __launch_bounds__(256) void val_kernel(
    const float* __restrict__ y, const float* __restrict__ Wv,
    const float* __restrict__ bv, float* __restrict__ out)
{
    __shared__ float red[8];
    const int b = blockIdx.x;
    const int t = threadIdx.x;
    float p = y[(size_t)b * PHN + t] * Wv[t];
    #pragma unroll
    for (int off = 16; off > 0; off >>= 1) p += __shfl_down(p, off, 32);
    if ((t & 31) == 0) red[t >> 5] = p;
    __syncthreads();
    if (t == 0) {
        float s = 0.0f;
        #pragma unroll
        for (int w = 0; w < 8; ++w) s += red[w];
        out[b] = s + bv[0];
    }
}

// -----------------------------------------------------------------------------
extern "C" void kernel_launch(void* const* d_in, const int* in_sizes, int n_in,
                              void* d_out, int out_size, void* d_ws, size_t ws_size,
                              hipStream_t stream)
{
    (void)in_sizes; (void)n_in; (void)out_size; (void)ws_size;
    const float* state = (const float*)d_in[0];
    const float* W1 = (const float*)d_in[1];
    const float* b1 = (const float*)d_in[2];
    const float* W2 = (const float*)d_in[3];
    const float* b2 = (const float*)d_in[4];
    const float* W3 = (const float*)d_in[5];
    const float* b3 = (const float*)d_in[6];
    const float* Wp = (const float*)d_in[7];
    const float* bp = (const float*)d_in[8];
    const float* Wv = (const float*)d_in[9];
    const float* bv = (const float*)d_in[10];

    // Workspace carve-up (256B aligned blocks)
    auto align256 = [](size_t x) { return (x + 255) & ~(size_t)255; };
    char* wsp = (char*)d_ws;
    __bf16* fcb = (__bf16*)wsp; wsp += align256((size_t)B_ * NIP * 2);   // 512 KB
    float*  yb  = (float*)wsp;  wsp += align256((size_t)B_ * PHN * 4);   // 256 KB
    __bf16* w1b = (__bf16*)wsp; wsp += align256((size_t)H1N * SEC_IN * 2);
    __bf16* w2b = (__bf16*)wsp; wsp += align256((size_t)H2N * H1N * 2);
    __bf16* wpb = (__bf16*)wsp; wsp += align256((size_t)PHN * NIP * 2);

    // Pre-convert weights to bf16 (Wp zero-padded to K=1024)
    {
        int t1 = H1N * SEC_IN;
        cvt_pad_kernel<<<(t1 + 255) / 256, 256, 0, stream>>>(W1, w1b, SEC_IN, SEC_IN, t1);
        int t2 = H2N * H1N;
        cvt_pad_kernel<<<(t2 + 255) / 256, 256, 0, stream>>>(W2, w2b, H1N, H1N, t2);
        int t3 = PHN * NIP;
        cvt_pad_kernel<<<(t3 + 255) / 256, 256, 0, stream>>>(Wp, wpb, NI, NIP, t3);
    }

    dim3 g1(B_, (NI + IT - 1) / IT);          // 256 x 16
    sec_mlp_kernel<<<g1, 256, 0, stream>>>(state, w1b, b1, w2b, b2, W3, b3, fcb);

    dim3 g2(B_ / 16, PHN / 16);               // 16 x 16
    pred_y_kernel<<<g2, 32, 0, stream>>>(fcb, wpb, bp, yb);

    val_kernel<<<B_, 256, 0, stream>>>(yb, Wv, bv, (float*)d_out);
}
